// FlashAttention_27539330302256
// MI455X (gfx1250) — matrix operand order repaired and checked by reading, not compiled, not run on hardware
//
#include <hip/hip_runtime.h>
#include <hip/hip_bf16.h>

typedef _Float16 v16h __attribute__((ext_vector_type(16)));
typedef _Float16 v8h  __attribute__((ext_vector_type(8)));
typedef float    v8f  __attribute__((ext_vector_type(8)));
typedef int      v4i_ __attribute__((vector_size(4 * sizeof(int))));
typedef __attribute__((address_space(3))) v4i_* lds_v4i_ptr;

namespace {
constexpr int kB = 2, kS = 8192, kD = 1024, kH = 16, kHD = 64;
// (1/sqrt(HD)) * log2(e): scores produced directly in log2 domain -> exp2
constexpr float kQScale = 0.125f * 1.4426950408889634f;
}

static __device__ __forceinline__ v8f wmma16(v16h a, v16h b, v8f c) {
  return __builtin_amdgcn_wmma_f32_16x16x32_f16(false, a, false, b, (short)0, c,
                                                false, false);
}
static __device__ __forceinline__ v16h mk16(v8h lo, v8h hi) {
  union { v8h h[2]; v16h v; } u;
  u.h[0] = lo; u.h[1] = hi;
  return u.v;
}
static __device__ __forceinline__ v8f zero8() {
  v8f z;
#pragma unroll
  for (int i = 0; i < 8; ++i) z[i] = 0.f;
  return z;
}
// exchange with lane^16 (wave32): ds_swizzle SWAPX16 (xor=0x10, and=0x1f)
static __device__ __forceinline__ float xswap16(float x) {
  return __int_as_float(__builtin_amdgcn_ds_swizzle(__float_as_int(x), 0x401F));
}

// ---------------------------------------------------------------------------
// Async global -> LDS copy of 32 bytes per lane (two b128 async ops).
// GLOBAL_LOAD_ASYNC_TO_LDS_* adds IOFFSET to both the global and LDS address,
// so offsets 0/16 cover both halves. Tracked with ASYNCcnt.
// Builtin signature (from compiler diagnostics): arg0 = int4-vector pointer
// (global source), arg1 = int4-vector __shared__ pointer (LDS dest),
// arg2 = imm offset, arg3 = imm cpol.
// ---------------------------------------------------------------------------
#if __has_builtin(__builtin_amdgcn_global_load_async_to_lds_b128) && \
    __has_builtin(__builtin_amdgcn_s_wait_asynccnt)
#define USE_ASYNC_LDS 1
#else
#define USE_ASYNC_LDS 0
#endif

#if USE_ASYNC_LDS
static __device__ __forceinline__ void cp32_g2l(const _Float16* g,
                                               _Float16* l) {
  __builtin_amdgcn_global_load_async_to_lds_b128(
      (v4i_*)(void*)g, (lds_v4i_ptr)(void*)l, 0, 0);
  __builtin_amdgcn_global_load_async_to_lds_b128(
      (v4i_*)(void*)g, (lds_v4i_ptr)(void*)l, 16, 0);
}
static __device__ __forceinline__ void cp_wait(int n) {
  if (n == 0)
    __builtin_amdgcn_s_wait_asynccnt(0);
  else if (n == 2)
    __builtin_amdgcn_s_wait_asynccnt(2);
  else
    __builtin_amdgcn_s_wait_asynccnt(4);
}
#else
static __device__ __forceinline__ void cp32_g2l(const _Float16* g,
                                               _Float16* l) {
  *(v8h*)l = *(const v8h*)g;
  *(v8h*)(l + 8) = *(const v8h*)(g + 8);
}
static __device__ __forceinline__ void cp_wait(int) {}
#endif

// ---------------------------------------------------------------------------
// Kernel 1: fp32 weights -> f16, transposed to [N][K] so GEMM B-operands are
// contiguous per lane. 4 matrices: wq, wk, wv, wo.
// ---------------------------------------------------------------------------
__global__ __launch_bounds__(256) void wprep_kernel(
    const float* __restrict__ wq, const float* __restrict__ wk,
    const float* __restrict__ wv, const float* __restrict__ wo,
    _Float16* __restrict__ wT) {
  __shared__ float tile[32][33];
  const int mat = blockIdx.z;
  const float* src = (mat == 0) ? wq : (mat == 1) ? wk : (mat == 2) ? wv : wo;
  _Float16* dst = wT + (size_t)mat * kD * kD;
  const int tx = threadIdx.x, ty = threadIdx.y;  // blockDim = (32, 8)
  const int n0 = blockIdx.x * 32, k0 = blockIdx.y * 32;
#pragma unroll
  for (int i = 0; i < 4; ++i)
    tile[ty + 8 * i][tx] = src[(size_t)(k0 + ty + 8 * i) * kD + n0 + tx];
  __syncthreads();
#pragma unroll
  for (int i = 0; i < 4; ++i)
    dst[(size_t)(n0 + ty + 8 * i) * kD + k0 + tx] =
        (_Float16)tile[tx][ty + 8 * i];
}

// ---------------------------------------------------------------------------
// Kernel 2: fused QKV projection GEMM, double-buffered.
//   C[16384 x 1024] = hidden(f32->f16) x w{q,k,v}T, 128x128 block tiles,
//   8 waves (2x4), each wave 4x2 WMMA subtiles, K-chunks of 32.
//   A tile: global fp32 prefetched 1 step ahead in regs, cvt + ds_store.
//   B tile: async global->LDS, 1 step ahead.
// Epilogue writes Q [B,H,S,HD] (pre-scaled), K [B,H,S,HD], V^T [B,H,HD,S].
// ---------------------------------------------------------------------------
__global__ __launch_bounds__(256) void qkv_kernel(
    const float* __restrict__ hid, const _Float16* __restrict__ wT,
    const float* __restrict__ bq, const float* __restrict__ bk,
    const float* __restrict__ bv, _Float16* __restrict__ Qo,
    _Float16* __restrict__ Ko, _Float16* __restrict__ VTo) {
  __shared__ _Float16 sA[2][128 * 40];  // 128 rows x 32 halves, stride 80B
  __shared__ _Float16 sB[2][128 * 40];  // B^T: 128 n-rows x 32 k-halves
  const int t = threadIdx.x;
  const int w = t >> 5, lane = t & 31, lg = lane >> 4, lm = lane & 15;
  const int wm = w & 1, wn = w >> 1;
  const int m0 = blockIdx.x * 128;
  const int proj = blockIdx.y >> 3;
  const int nc = (blockIdx.y & 7) * 128;
  const _Float16* wp = wT + (size_t)proj * kD * kD;
  const float* bias = (proj == 0) ? bq : (proj == 1) ? bk : bv;

  v8f acc[4][2];
#pragma unroll
  for (int i = 0; i < 4; ++i)
#pragma unroll
    for (int j = 0; j < 2; ++j) acc[i][j] = zero8();

  const int ar = t >> 1, acg = (t & 1) * 16;  // staging: row, 16-elem group

  float4 fr[4];  // A prefetch registers (one k-step ahead)
  auto loadA = [&](int k0) {
    const float* gp = hid + (size_t)(m0 + ar) * kD + k0 + acg;
    fr[0] = ((const float4*)gp)[0];
    fr[1] = ((const float4*)gp)[1];
    fr[2] = ((const float4*)gp)[2];
    fr[3] = ((const float4*)gp)[3];
  };
  auto storeA = [&](int buf) {
    v8h h0, h1;
    h0[0] = (_Float16)fr[0].x; h0[1] = (_Float16)fr[0].y;
    h0[2] = (_Float16)fr[0].z; h0[3] = (_Float16)fr[0].w;
    h0[4] = (_Float16)fr[1].x; h0[5] = (_Float16)fr[1].y;
    h0[6] = (_Float16)fr[1].z; h0[7] = (_Float16)fr[1].w;
    h1[0] = (_Float16)fr[2].x; h1[1] = (_Float16)fr[2].y;
    h1[2] = (_Float16)fr[2].z; h1[3] = (_Float16)fr[2].w;
    h1[4] = (_Float16)fr[3].x; h1[5] = (_Float16)fr[3].y;
    h1[6] = (_Float16)fr[3].z; h1[7] = (_Float16)fr[3].w;
    *(v8h*)&sA[buf][ar * 40 + acg] = h0;
    *(v8h*)&sA[buf][ar * 40 + acg + 8] = h1;
  };
  auto issueB = [&](int k0, int buf) {
    cp32_g2l(wp + (size_t)(nc + ar) * kD + k0 + acg, &sB[buf][ar * 40 + acg]);
  };

  loadA(0);
  issueB(0, 0);
  for (int stp = 0; stp < 32; ++stp) {
    const int bsel = stp & 1;
    const bool more = (stp + 1) < 32;
    storeA(bsel);  // A(stp): regs loaded one step ago
    if (more) {
      loadA(32 * (stp + 1));
      issueB(32 * (stp + 1), bsel ^ 1);  // safe: buf^1 last read at stp-1
    }
    cp_wait(more ? 2 : 0);  // B(stp) landed (in-order ASYNCcnt)
    __syncthreads();

    v16h af[4], bf[2];
#pragma unroll
    for (int i = 0; i < 4; ++i) {
      const int row = wm * 64 + i * 16 + lm;
      af[i] = mk16(*(const v8h*)&sA[bsel][row * 40 + 16 * lg],
                   *(const v8h*)&sA[bsel][row * 40 + 16 * lg + 8]);
    }
#pragma unroll
    for (int j = 0; j < 2; ++j) {
      const int row = wn * 32 + j * 16 + lm;
      bf[j] = mk16(*(const v8h*)&sB[bsel][row * 40 + 16 * lg],
                   *(const v8h*)&sB[bsel][row * 40 + 16 * lg + 8]);
    }
#pragma unroll
    for (int i = 0; i < 4; ++i)
#pragma unroll
      for (int j = 0; j < 2; ++j) acc[i][j] = wmma16(af[i], bf[j], acc[i][j]);
    __syncthreads();
  }

  const int b = m0 / kS;                  // uniform per block (S % 128 == 0)
  const int sbase = (m0 % kS) + wm * 64;  // sequence position base
#pragma unroll
  for (int j = 0; j < 2; ++j) {
    const int col = nc + wn * 32 + j * 16 + lm;  // 0..1023
    const int hh = col >> 6, hd = col & 63;
    const float bsv = bias[col];
#pragma unroll
    for (int i = 0; i < 4; ++i) {
      const int srow = sbase + i * 16 + 8 * lg;  // 8 consecutive s per lane
      if (proj == 2) {  // V^T [B,H,HD,S]: pack 8 f16 along s -> one b128
        v8h pk;
#pragma unroll
        for (int r = 0; r < 8; ++r) pk[r] = (_Float16)(acc[i][j][r] + bsv);
        *(v8h*)&VTo[((size_t)(b * kH + hh) * kHD + hd) * kS + srow] = pk;
      } else if (proj == 0) {  // Q [B,H,S,HD], pre-scaled into log2 domain
        _Float16* dst = Qo + ((size_t)(b * kH + hh) * kS + srow) * kHD + hd;
#pragma unroll
        for (int r = 0; r < 8; ++r)
          dst[(size_t)r * kHD] = (_Float16)((acc[i][j][r] + bsv) * kQScale);
      } else {  // K [B,H,S,HD]
        _Float16* dst = Ko + ((size_t)(b * kH + hh) * kS + srow) * kHD + hd;
#pragma unroll
        for (int r = 0; r < 8; ++r)
          dst[(size_t)r * kHD] = (_Float16)(acc[i][j][r] + bsv);
      }
    }
  }
}

// ---------------------------------------------------------------------------
// Kernel 3: causal flash attention, double-buffered async K/V^T staging.
// Block = (qb, h, b): 128 q rows, 8 waves, wave owns 16 q rows.
// S^T = K·Q^T (softmax per-lane), O^T = V^T·P^T. 64-key LDS tiles, 32-key
// WMMA chunks; chunks entirely above the wave diagonal are skipped.
// ---------------------------------------------------------------------------
__global__ __launch_bounds__(256) void attn_kernel(
    const _Float16* __restrict__ Q, const _Float16* __restrict__ K,
    const _Float16* __restrict__ VT, _Float16* __restrict__ attn) {
  __shared__ _Float16 sK[2][64 * 72];     // 64 keys x 64 hd, stride 144B
  __shared__ _Float16 sV[2][64 * 72];     // 64 hd x 64 keys
  __shared__ _Float16 sP[8 * 16 * 40];    // per-wave P^T scratch, stride 80B
  const int t = threadIdx.x;
  const int w = t >> 5, lane = t & 31, lg = lane >> 4, lm = lane & 15;
  const int qb = blockIdx.x, hh = blockIdx.y, b = blockIdx.z;
  const _Float16* Qb = Q + (size_t)(b * kH + hh) * kS * kHD;
  const _Float16* Kb = K + (size_t)(b * kH + hh) * kS * kHD;
  const _Float16* Vb = VT + (size_t)(b * kH + hh) * kHD * kS;
  const int q0 = qb * 128 + w * 16;

  v16h qf[2];  // Q^T B-operands: contiguous 32B per lane, loaded once
#pragma unroll
  for (int c = 0; c < 2; ++c) {
    const _Float16* qp = Qb + (size_t)(q0 + lm) * kHD + 32 * c + 16 * lg;
    qf[c] = mk16(*(const v8h*)qp, *(const v8h*)(qp + 8));
  }

  float mi = -__builtin_inff();
  float li = 0.f;
  v8f o[4];
#pragma unroll
  for (int ht = 0; ht < 4; ++ht) o[ht] = zero8();
  _Float16* Pw = &sP[w * 640];
  const int kend = qb * 128 + 128;
  const int sr = t >> 2, sc = (t & 3) * 16;

  auto stageKV = [&](int j0, int buf) {
    cp32_g2l(Kb + (size_t)(j0 + sr) * kHD + sc, &sK[buf][sr * 72 + sc]);
    cp32_g2l(Vb + (size_t)sr * kS + j0 + sc, &sV[buf][sr * 72 + sc]);
  };

  const int nsteps = kend / 64;
  stageKV(0, 0);
  for (int stp = 0; stp < nsteps; ++stp) {
    const int j0 = stp * 64, bsel = stp & 1;
    const bool more = (stp + 1) < nsteps;
    if (more) stageKV(j0 + 64, bsel ^ 1);  // buf^1 last read at stp-1
    cp_wait(more ? 4 : 0);                 // this step's 4 copies landed
    __syncthreads();
#pragma unroll
    for (int s2 = 0; s2 < 2; ++s2) {
      const int kb = j0 + 32 * s2;
      if (kb < q0 + 16) {  // wave-uniform causal skip
        v8f st[2];
        st[0] = zero8();
        st[1] = zero8();
#pragma unroll
        for (int kt = 0; kt < 2; ++kt) {  // S^T = K (A) x Q^T (B)
          const int kr = 32 * s2 + kt * 16 + lm;
#pragma unroll
          for (int c = 0; c < 2; ++c) {
            v16h ak =
                mk16(*(const v8h*)&sK[bsel][kr * 72 + 32 * c + 16 * lg],
                     *(const v8h*)&sK[bsel][kr * 72 + 32 * c + 16 * lg + 8]);
            st[kt] = wmma16(ak, qf[c], st[kt]);
          }
        }
        const int q = q0 + lm;  // this lane's q row (n-column of S^T)
        float lmax = -3.0e38f;
#pragma unroll
        for (int kt = 0; kt < 2; ++kt)
#pragma unroll
          for (int r = 0; r < 8; ++r) {
            const int key = kb + kt * 16 + 8 * lg + r;
            float sv = st[kt][r];
            sv = (key > q) ? -1.0e30f : sv;  // causal mask
            st[kt][r] = sv;
            lmax = fmaxf(lmax, sv);
          }
        const float cmax = fmaxf(lmax, xswap16(lmax));  // join lane groups
        const float mnew = fmaxf(mi, cmax);
        const float rescale = __builtin_amdgcn_exp2f(mi - mnew);
        float lsum = 0.f;
#pragma unroll
        for (int kt = 0; kt < 2; ++kt) {  // P^T to per-wave LDS scratch
          v8h ph;
#pragma unroll
          for (int r = 0; r < 8; ++r) {
            const float p = __builtin_amdgcn_exp2f(st[kt][r] - mnew);
            lsum += p;
            ph[r] = (_Float16)p;
          }
          *(v8h*)&Pw[lm * 40 + kt * 16 + 8 * lg] = ph;
        }
        li = li * rescale + lsum + xswap16(lsum);
#pragma unroll
        for (int ht = 0; ht < 4; ++ht)
#pragma unroll
          for (int r = 0; r < 8; ++r) o[ht][r] *= rescale;
        v16h pb = mk16(*(const v8h*)&Pw[lm * 40 + 16 * lg],
                       *(const v8h*)&Pw[lm * 40 + 16 * lg + 8]);
#pragma unroll
        for (int ht = 0; ht < 4; ++ht) {  // O^T += V^T (A) x P^T (B)
          const int vr = ht * 16 + lm;
          v16h av =
              mk16(*(const v8h*)&sV[bsel][vr * 72 + 32 * s2 + 16 * lg],
                   *(const v8h*)&sV[bsel][vr * 72 + 32 * s2 + 16 * lg + 8]);
          o[ht] = wmma16(av, pb, o[ht]);
        }
        mi = mnew;
      }
    }
    __syncthreads();
  }

  const float inv = 1.0f / li;
  _Float16* op = attn + (size_t)(b * kS + q0 + lm) * kD + hh * kHD;
#pragma unroll
  for (int ht = 0; ht < 4; ++ht) {  // lane holds hd = ht*16 + 8lg + r at q=lm
    v8h pk;
#pragma unroll
    for (int r = 0; r < 8; ++r) pk[r] = (_Float16)(o[ht][r] * inv);
    *(v8h*)&op[ht * 16 + 8 * lg] = pk;
  }
}

// ---------------------------------------------------------------------------
// Kernel 4: out projection: out = attn(f16) x wo(f16,T) + bo, fp32 output.
// Fully async double-buffered staging (both tiles already f16).
// ---------------------------------------------------------------------------
__global__ __launch_bounds__(256) void oproj_kernel(
    const _Float16* __restrict__ A, const _Float16* __restrict__ Wt,
    const float* __restrict__ bo, float* __restrict__ out) {
  __shared__ _Float16 sA[2][128 * 40];
  __shared__ _Float16 sB[2][128 * 40];
  const int t = threadIdx.x;
  const int w = t >> 5, lane = t & 31, lg = lane >> 4, lm = lane & 15;
  const int wm = w & 1, wn = w >> 1;
  const int m0 = blockIdx.x * 128;
  const int nc = blockIdx.y * 128;

  v8f acc[4][2];
#pragma unroll
  for (int i = 0; i < 4; ++i)
#pragma unroll
    for (int j = 0; j < 2; ++j) acc[i][j] = zero8();

  const int ar = t >> 1, acg = (t & 1) * 16;
  auto issueAB = [&](int k0, int buf) {
    cp32_g2l(A + (size_t)(m0 + ar) * kD + k0 + acg, &sA[buf][ar * 40 + acg]);
    cp32_g2l(Wt + (size_t)(nc + ar) * kD + k0 + acg, &sB[buf][ar * 40 + acg]);
  };

  issueAB(0, 0);
  for (int stp = 0; stp < 32; ++stp) {
    const int bsel = stp & 1;
    const bool more = (stp + 1) < 32;
    if (more) issueAB(32 * (stp + 1), bsel ^ 1);
    cp_wait(more ? 4 : 0);
    __syncthreads();

    v16h af[4], bf[2];
#pragma unroll
    for (int i = 0; i < 4; ++i) {
      const int row = wm * 64 + i * 16 + lm;
      af[i] = mk16(*(const v8h*)&sA[bsel][row * 40 + 16 * lg],
                   *(const v8h*)&sA[bsel][row * 40 + 16 * lg + 8]);
    }
#pragma unroll
    for (int j = 0; j < 2; ++j) {
      const int row = wn * 32 + j * 16 + lm;
      bf[j] = mk16(*(const v8h*)&sB[bsel][row * 40 + 16 * lg],
                   *(const v8h*)&sB[bsel][row * 40 + 16 * lg + 8]);
    }
#pragma unroll
    for (int i = 0; i < 4; ++i)
#pragma unroll
      for (int j = 0; j < 2; ++j) acc[i][j] = wmma16(af[i], bf[j], acc[i][j]);
    __syncthreads();
  }

#pragma unroll
  for (int j = 0; j < 2; ++j) {
    const int col = nc + wn * 32 + j * 16 + lm;
    const float bov = bo[col];
#pragma unroll
    for (int i = 0; i < 4; ++i) {
      const int row = m0 + wm * 64 + i * 16 + 8 * lg;
#pragma unroll
      for (int r = 0; r < 8; ++r)
        out[(size_t)(row + r) * kD + col] = acc[i][j][r] + bov;
    }
  }
}

// ---------------------------------------------------------------------------
extern "C" void kernel_launch(void* const* d_in, const int* in_sizes, int n_in,
                              void* d_out, int out_size, void* d_ws,
                              size_t ws_size, hipStream_t stream) {
  (void)in_sizes; (void)n_in; (void)out_size; (void)ws_size;
  const float* hid = (const float*)d_in[0];
  const float* wq = (const float*)d_in[1];
  const float* bq = (const float*)d_in[2];
  const float* wk = (const float*)d_in[3];
  const float* bk = (const float*)d_in[4];
  const float* wv = (const float*)d_in[5];
  const float* bv = (const float*)d_in[6];
  const float* wo = (const float*)d_in[7];
  const float* bo = (const float*)d_in[8];

  char* ws = (char*)d_ws;
  const size_t MB = 1024 * 1024;
  _Float16* wT = (_Float16*)ws;                    // 4 x 2 MB (wqT,wkT,wvT,woT)
  _Float16* Qb = (_Float16*)(ws + 8 * MB);         // 32 MB [B,H,S,HD] f16
  _Float16* Kb = (_Float16*)(ws + 40 * MB);        // 32 MB [B,H,S,HD] f16
  _Float16* Vb = (_Float16*)(ws + 72 * MB);        // 32 MB [B,H,HD,S] f16
  _Float16* At = (_Float16*)(ws + 104 * MB);       // 32 MB [B*S, H*HD] f16

  wprep_kernel<<<dim3(32, 32, 4), dim3(32, 8), 0, stream>>>(wq, wk, wv, wo, wT);
  qkv_kernel<<<dim3((kB * kS) / 128, 24), 256, 0, stream>>>(
      hid, wT, bq, bk, bv, Qb, Kb, Vb);
  attn_kernel<<<dim3(kS / 128, kH, kB), 256, 0, stream>>>(Qb, Kb, Vb, At);
  oproj_kernel<<<dim3((kB * kS) / 128, kD / 128), 256, 0, stream>>>(
      At, wT + (size_t)3 * kD * kD, bo, (float*)d_out);
}
